// PointerNetwork_28260884807952
// MI455X (gfx1250) — matrix operand order repaired and checked by reading, not compiled, not run on hardware
//
#include <hip/hip_runtime.h>
#include <hip/hip_bf16.h>
#include <math.h>

// ---------------------------------------------------------------------------
// Problem constants (from reference): B=64, S=256, E=256, H=512, 4H=2048
// ---------------------------------------------------------------------------
#define BN   64
#define SN   256
#define EN   256
#define HN   512
#define G4H  2048

typedef __bf16 bf16_t;
typedef bf16_t v16bf __attribute__((ext_vector_type(16)));
typedef float  v8f   __attribute__((ext_vector_type(8)));

// fp32 -> bf16 (round to nearest even), stored as raw u16
static __device__ __forceinline__ unsigned short f2bf(float f) {
    unsigned int u = __float_as_uint(f);
    u += 0x7FFFu + ((u >> 16) & 1u);
    return (unsigned short)(u >> 16);
}

static __device__ __forceinline__ float sigmoidf(float x) {
    return 1.0f / (1.0f + expf(-x));
}

// A-fragment (16x32 bf16) K offset within a 32-chunk for (lane, e):
//  lanes 0-15 : e=0..7 -> K=0..7,   e=8..15 -> K=16..23
//  lanes 16-31: e=0..7 -> K=8..15,  e=8..15 -> K=24..31
static __device__ __forceinline__ int koffA(int lane, int e) {
    return ((e < 8) ? e : (e + 8)) + ((lane < 16) ? 0 : 8);
}

// Inverse: given row-in-tile m (=b%16) and K offset ko (0..31), find (lane,e)
static __device__ __forceinline__ void invA(int m, int ko, int& lane, int& e) {
    if (ko < 8)       { lane = m;      e = ko;      }
    else if (ko < 16) { lane = m + 16; e = ko - 8;  }
    else if (ko < 24) { lane = m;      e = ko - 8;  }
    else              { lane = m + 16; e = ko - 16; }
}

// ---------------------------------------------------------------------------
// Weight packing: B-fragment (32x16 bf16 per K-chunk) of W^T, i.e.
// Bfrag[ntile][kc][lane*16+e] = bf16( Wsrc[n][k] ), n = ntile*16 + (lane&15),
// k = kc*32 + (e<8 ? e : e+8) + (lane<16 ? 0 : 8).  K is the concat [Ka | Kb] of two mats.
// ---------------------------------------------------------------------------
__global__ void pack_w(const float* __restrict__ Wa, int Ka,
                       const float* __restrict__ Wb, int Kb,
                       int Nout, unsigned short* __restrict__ dst)
{
    const int Ktot = Ka + Kb;
    const int kcCount = Ktot >> 5;
    const size_t total = (size_t)(Nout >> 4) * kcCount * 512;
    for (size_t t = (size_t)blockIdx.x * blockDim.x + threadIdx.x; t < total;
         t += (size_t)gridDim.x * blockDim.x) {
        int e     = (int)(t & 15);
        int lane  = (int)((t >> 4) & 31);
        int kc    = (int)((t >> 9) % kcCount);
        int ntile = (int)(t / ((size_t)kcCount << 9));
        int n = ntile * 16 + (lane & 15);
        int k = kc * 32 + ((e < 8) ? e : (e + 8)) + ((lane < 16) ? 0 : 8);
        float v = (k < Ka) ? Wa[(size_t)n * Ka + k]
                           : Wb[(size_t)n * Kb + (k - Ka)];
        dst[t] = f2bf(v);
    }
}

// ---------------------------------------------------------------------------
// Embedding lookup -> encoder x A-fragments, for every timestep.
// xfrag[((s*4 + mtile)*8 + kc)*512 + lane*16 + e] = bf16(emb[src[b][s]][K])
// ---------------------------------------------------------------------------
__global__ void embed_pack(const int* __restrict__ src,
                           const float* __restrict__ emb,
                           unsigned short* __restrict__ xfrag)
{
    const size_t total = (size_t)SN * 4 * 8 * 512;   // 4,194,304
    for (size_t t = (size_t)blockIdx.x * blockDim.x + threadIdx.x; t < total;
         t += (size_t)gridDim.x * blockDim.x) {
        int e     = (int)(t & 15);
        int lane  = (int)((t >> 4) & 31);
        int kc    = (int)((t >> 9) & 7);
        int mtile = (int)((t >> 12) & 3);
        int s     = (int)(t >> 14);
        int b = mtile * 16 + (lane & 15);
        int K = kc * 32 + koffA(lane, e);
        int tok = src[b * SN + s];
        xfrag[t] = f2bf(emb[(size_t)tok * EN + K]);
    }
}

__global__ void valid_kernel(const int* __restrict__ src_len, int* __restrict__ out_valid)
{
    int b = blockIdx.x, s = threadIdx.x;
    out_valid[b * SN + s] = (s < src_len[b]) ? 1 : 0;
}

__global__ void init_state(float* h, float* c,
                           unsigned short* hfrag, unsigned short* xinfrag)
{
    int t = blockIdx.x * blockDim.x + threadIdx.x;
    if (t < BN * HN) { h[t] = 0.f; c[t] = 0.f; }
    if (t < 4 * 16 * 512) { hfrag[t] = 0; xinfrag[t] = 0; }
}

// ---------------------------------------------------------------------------
// WMMA GEMM with M-direction register blocking:
//   C[M x Nout] = [A0 | A1] (bf16 A-frags) x Bw (bf16 B-frags) + bias
// Each wave owns ONE ntile and FOUR consecutive mtiles (M-group of 64 rows):
// the B fragment is loaded once per K-chunk and reused by 4 independent
// WMMAs (4 accumulators), quartering weight traffic from L2 and giving the
// SIMD 4 independent v_wmma chains to pipeline.
// blockDim = (32,4); gridDim = (Mtiles/4, Ntiles/4).
// ---------------------------------------------------------------------------
__global__ void gemm_wmma(const unsigned short* __restrict__ a0, int kc0,
                          const unsigned short* __restrict__ a1, int kc1,
                          const unsigned short* __restrict__ bw,
                          const float* __restrict__ bias,
                          float* __restrict__ C, int Nout)
{
    const int mg    = blockIdx.x;                          // group of 4 mtiles
    const int ntile = blockIdx.y * blockDim.y + threadIdx.y;
    const int lane  = threadIdx.x;
    const int kcTot = kc0 + kc1;

    v8f acc0, acc1, acc2, acc3;
    const float binit = bias ? bias[ntile * 16 + (lane & 15)] : 0.0f;
#pragma unroll
    for (int r = 0; r < 8; ++r) { acc0[r] = binit; acc1[r] = binit; acc2[r] = binit; acc3[r] = binit; }

    const unsigned short* bp = bw + ((size_t)ntile * kcTot) * 512 + lane * 16;
    const size_t aStride0 = (size_t)kc0 * 512;             // per-mtile stride in a0
    const size_t aStride1 = (size_t)kc1 * 512;
    const unsigned short* ap0 = a0 + (size_t)(mg * 4) * aStride0 + lane * 16;

    for (int kc = 0; kc < kc0; ++kc) {
        __builtin_prefetch(bp + (size_t)(kc + 1) * 512, 0, 0);   // global_prefetch_b8
        v16bf bv = *(const v16bf*)(bp + (size_t)kc * 512);
        v16bf av0 = *(const v16bf*)(ap0 + 0 * aStride0 + (size_t)kc * 512);
        v16bf av1 = *(const v16bf*)(ap0 + 1 * aStride0 + (size_t)kc * 512);
        v16bf av2 = *(const v16bf*)(ap0 + 2 * aStride0 + (size_t)kc * 512);
        v16bf av3 = *(const v16bf*)(ap0 + 3 * aStride0 + (size_t)kc * 512);
        acc0 = __builtin_amdgcn_wmma_f32_16x16x32_bf16(false, av0, false, bv, (short)0, acc0, false, false);
        acc1 = __builtin_amdgcn_wmma_f32_16x16x32_bf16(false, av1, false, bv, (short)0, acc1, false, false);
        acc2 = __builtin_amdgcn_wmma_f32_16x16x32_bf16(false, av2, false, bv, (short)0, acc2, false, false);
        acc3 = __builtin_amdgcn_wmma_f32_16x16x32_bf16(false, av3, false, bv, (short)0, acc3, false, false);
    }
    if (a1) {
        const unsigned short* ap1 = a1 + (size_t)(mg * 4) * aStride1 + lane * 16;
        const unsigned short* bp1 = bp + (size_t)kc0 * 512;
        for (int kc = 0; kc < kc1; ++kc) {
            __builtin_prefetch(bp1 + (size_t)(kc + 1) * 512, 0, 0);
            v16bf bv = *(const v16bf*)(bp1 + (size_t)kc * 512);
            v16bf av0 = *(const v16bf*)(ap1 + 0 * aStride1 + (size_t)kc * 512);
            v16bf av1 = *(const v16bf*)(ap1 + 1 * aStride1 + (size_t)kc * 512);
            v16bf av2 = *(const v16bf*)(ap1 + 2 * aStride1 + (size_t)kc * 512);
            v16bf av3 = *(const v16bf*)(ap1 + 3 * aStride1 + (size_t)kc * 512);
            acc0 = __builtin_amdgcn_wmma_f32_16x16x32_bf16(false, av0, false, bv, (short)0, acc0, false, false);
            acc1 = __builtin_amdgcn_wmma_f32_16x16x32_bf16(false, av1, false, bv, (short)0, acc1, false, false);
            acc2 = __builtin_amdgcn_wmma_f32_16x16x32_bf16(false, av2, false, bv, (short)0, acc2, false, false);
            acc3 = __builtin_amdgcn_wmma_f32_16x16x32_bf16(false, av3, false, bv, (short)0, acc3, false, false);
        }
    }
    // D layout: VGPR r -> row (mtile*16 + r) for lanes 0-15, row (+8) for 16-31
    const int col = ntile * 16 + (lane & 15);
    const int rb  = (lane < 16) ? 0 : 8;
#pragma unroll
    for (int r = 0; r < 8; ++r) {
        C[(size_t)((mg * 4 + 0) * 16 + rb + r) * Nout + col] = acc0[r];
        C[(size_t)((mg * 4 + 1) * 16 + rb + r) * Nout + col] = acc1[r];
        C[(size_t)((mg * 4 + 2) * 16 + rb + r) * Nout + col] = acc2[r];
        C[(size_t)((mg * 4 + 3) * 16 + rb + r) * Nout + col] = acc3[r];
    }
}

// ---------------------------------------------------------------------------
// Encoder LSTM cell (masked) + write enc_out, h-frag (next step), enc-frag
// (for the hoisted W1 projection GEMM). Gate order i,f,g,o (torch).
// ---------------------------------------------------------------------------
__global__ void lstm_cell_enc(const float* __restrict__ G,
                              float* __restrict__ h, float* __restrict__ c,
                              const int* __restrict__ src_len, int t,
                              float* __restrict__ enc_out,
                              unsigned short* __restrict__ hfrag,
                              unsigned short* __restrict__ encfrag)
{
    int tid = blockIdx.x * blockDim.x + threadIdx.x;
    if (tid >= BN * HN) return;
    int b = tid >> 9, j = tid & (HN - 1);
    const float* g = G + (size_t)b * G4H;
    float ig = sigmoidf(g[j]);
    float fg = sigmoidf(g[HN + j]);
    float gg = tanhf(g[2 * HN + j]);
    float og = sigmoidf(g[3 * HN + j]);
    float c_old = c[tid], h_old = h[tid];
    float cn = fg * c_old + ig * gg;
    float hn = og * tanhf(cn);
    bool m = (t < src_len[b]);
    float hv = m ? hn : h_old;
    float cv = m ? cn : c_old;
    h[tid] = hv; c[tid] = cv;
    float eo = m ? hv : 0.0f;
    enc_out[((size_t)b * SN + t) * HN + j] = eo;

    int lane, e;
    invA(b & 15, j & 31, lane, e);
    hfrag[(((b >> 4) * 16) + (j >> 5)) * 512 + lane * 16 + e] = f2bf(hv);
    encfrag[((((size_t)t * 4 + (b >> 4)) * 16) + (j >> 5)) * 512 + lane * 16 + e] = f2bf(eo);
}

// Decoder LSTM cell (unmasked) + h-frag (feeds W2 GEMM and next-step GEMM)
__global__ void lstm_cell_dec(const float* __restrict__ G,
                              float* __restrict__ h, float* __restrict__ c,
                              unsigned short* __restrict__ hfrag)
{
    int tid = blockIdx.x * blockDim.x + threadIdx.x;
    if (tid >= BN * HN) return;
    int b = tid >> 9, j = tid & (HN - 1);
    const float* g = G + (size_t)b * G4H;
    float ig = sigmoidf(g[j]);
    float fg = sigmoidf(g[HN + j]);
    float gg = tanhf(g[2 * HN + j]);
    float og = sigmoidf(g[3 * HN + j]);
    float cn = fg * c[tid] + ig * gg;
    float hn = og * tanhf(cn);
    h[tid] = hn; c[tid] = cn;
    int lane, e;
    invA(b & 15, j & 31, lane, e);
    hfrag[(((b >> 4) * 16) + (j >> 5)) * 512 + lane * 16 + e] = f2bf(hn);
}

// ---------------------------------------------------------------------------
// Attention score: score[b][s] = sum_j tanh(enc_proj[s][b][j] + hw2[b][j])*vt[j]
// One wave per (b,s) pair.
// ---------------------------------------------------------------------------
__global__ void attn_score(const float* __restrict__ enc_proj,
                           const float* __restrict__ hw2,
                           const float* __restrict__ vt,
                           float* __restrict__ score)
{
    int w = (blockIdx.x * blockDim.x + threadIdx.x) >> 5;
    int lane = threadIdx.x & 31;
    if (w >= BN * SN) return;
    int b = w >> 8, s = w & (SN - 1);
    const float* ep = enc_proj + ((size_t)s * BN + b) * HN;  // (S,B,H) layout
    const float* hb = hw2 + (size_t)b * HN;
    float sum = 0.0f;
#pragma unroll
    for (int k = 0; k < HN / 32; ++k) {
        int j = lane + 32 * k;
        sum += tanhf(ep[j] + hb[j]) * vt[j];
    }
#pragma unroll
    for (int o = 16; o; o >>= 1) sum += __shfl_xor(sum, o, 32);
    if (lane == 0) score[w] = sum;
}

// ---------------------------------------------------------------------------
// Masked log-softmax over S, first-max argmax, write outputs, and pack the
// gathered enc_out row as the next decoder-input A-fragment.
// One block per batch row b; 256 threads (one per s).
// ---------------------------------------------------------------------------
__global__ void softmax_pack(const float* __restrict__ score,
                             const int* __restrict__ src_len, int step,
                             float* __restrict__ out_scores,
                             int* __restrict__ out_idx,
                             const float* __restrict__ enc_out,
                             unsigned short* __restrict__ xinfrag)
{
    __shared__ float red[SN];
    __shared__ int   redi[SN];
    __shared__ int   sidx;
    const int b = blockIdx.x, s = threadIdx.x;
    const int len = src_len[b];
    const float sc = score[b * SN + s];
    const bool  mm = (s < len) && (step < len);
    const float v = sc + (mm ? 0.0f : logf(1e-45f));   // log(m + 1e-45)

    red[s] = v; __syncthreads();
    for (int o = 128; o; o >>= 1) { if (s < o) red[s] = fmaxf(red[s], red[s + o]); __syncthreads(); }
    const float mx = red[0]; __syncthreads();
    red[s] = expf(v - mx); __syncthreads();
    for (int o = 128; o; o >>= 1) { if (s < o) red[s] += red[s + o]; __syncthreads(); }
    const float lse = logf(red[0]) + mx; __syncthreads();
    out_scores[((size_t)b * SN + step) * SN + s] = v - lse;

    // argmax over masked raw score (equivalent ordering; first max on ties)
    red[s] = mm ? sc : -INFINITY; redi[s] = s; __syncthreads();
    for (int o = 128; o; o >>= 1) {
        if (s < o) {
            bool take = (red[s + o] > red[s]) ||
                        (red[s + o] == red[s] && redi[s + o] < redi[s]);
            if (take) { red[s] = red[s + o]; redi[s] = redi[s + o]; }
        }
        __syncthreads();
    }
    if (s == 0) { sidx = redi[0]; out_idx[b * SN + step] = redi[0]; }
    __syncthreads();
    const int idx = sidx;

    // pack xnext = enc_out[b][idx][:] into decoder-input A-fragments
    for (int j = s; j < HN; j += SN) {
        float x = enc_out[((size_t)b * SN + idx) * HN + j];
        int lane, e;
        invA(b & 15, j & 31, lane, e);
        xinfrag[(((b >> 4) * 16) + (j >> 5)) * 512 + lane * 16 + e] = f2bf(x);
    }
}

// ---------------------------------------------------------------------------
// Host-side orchestration
// ---------------------------------------------------------------------------
extern "C" void kernel_launch(void* const* d_in, const int* in_sizes, int n_in,
                              void* d_out, int out_size, void* d_ws, size_t ws_size,
                              hipStream_t stream)
{
    (void)in_sizes; (void)n_in; (void)out_size; (void)ws_size;
    const int*   src     = (const int*)  d_in[0];
    const int*   src_len = (const int*)  d_in[1];
    const float* emb     = (const float*)d_in[2];
    const float* W_ih_e  = (const float*)d_in[3];
    const float* W_hh_e  = (const float*)d_in[4];
    const float* b_e     = (const float*)d_in[5];
    const float* W_ih_d  = (const float*)d_in[6];
    const float* W_hh_d  = (const float*)d_in[7];
    const float* b_d     = (const float*)d_in[8];
    const float* W1      = (const float*)d_in[9];
    const float* W2      = (const float*)d_in[10];
    const float* vt      = (const float*)d_in[11];

    // Workspace layout (all 256B-aligned sizes)
    char* ws = (char*)d_ws;
    size_t off = 0;
    auto take = [&](size_t bytes) { char* p = ws + off; off += (bytes + 255) & ~(size_t)255; return p; };
    unsigned short* encWf   = (unsigned short*)take((size_t)128 * 24 * 512 * 2); // [Wih_e;Whh_e]^T frags
    unsigned short* decWf   = (unsigned short*)take((size_t)128 * 32 * 512 * 2);
    unsigned short* W1f     = (unsigned short*)take((size_t)32 * 16 * 512 * 2);
    unsigned short* W2f     = (unsigned short*)take((size_t)32 * 16 * 512 * 2);
    unsigned short* xfrag   = (unsigned short*)take((size_t)SN * 4 * 8 * 512 * 2);
    unsigned short* hfrag   = (unsigned short*)take((size_t)4 * 16 * 512 * 2);
    unsigned short* xinfrag = (unsigned short*)take((size_t)4 * 16 * 512 * 2);
    float* h        = (float*)take((size_t)BN * HN * 4);
    float* c        = (float*)take((size_t)BN * HN * 4);
    float* gates    = (float*)take((size_t)BN * G4H * 4);
    float* enc_out  = (float*)take((size_t)BN * SN * HN * 4);      // (B,S,H)
    unsigned short* encfrag = (unsigned short*)take((size_t)SN * 4 * 16 * 512 * 2);
    float* enc_proj = (float*)take((size_t)SN * BN * HN * 4);      // (S,B,H)
    float* hw2      = (float*)take((size_t)BN * HN * 4);
    float* score    = (float*)take((size_t)BN * SN * 4);

    float* out_scores = (float*)d_out;                       // (B,S,S) f32
    int*   out_idx    = (int*)(out_scores + (size_t)BN * SN * SN);
    int*   out_valid  = out_idx + BN * SN;

    // One-time preparation
    init_state<<<dim3((BN * HN + 255) / 256), dim3(256), 0, stream>>>(h, c, hfrag, xinfrag);
    pack_w<<<dim3(4096), dim3(256), 0, stream>>>(W_ih_e, EN, W_hh_e, HN, G4H, encWf);
    pack_w<<<dim3(4096), dim3(256), 0, stream>>>(W_ih_d, HN, W_hh_d, HN, G4H, decWf);
    pack_w<<<dim3(1024), dim3(256), 0, stream>>>(W1, HN, W1, 0, HN, W1f);
    pack_w<<<dim3(1024), dim3(256), 0, stream>>>(W2, HN, W2, 0, HN, W2f);
    embed_pack<<<dim3(4096), dim3(256), 0, stream>>>(src, emb, xfrag);
    valid_kernel<<<dim3(BN), dim3(SN), 0, stream>>>(src_len, out_valid);

    const dim3 wblk(32, 4);   // 4 waves per block, one ntile each

    // Encoder: per step, gates = [x_t, h] @ [Wih;Whh]^T + b  ->  cell update
    // grid = (Mtiles/4, Ntiles/4) = (1, 32)
    for (int t = 0; t < SN; ++t) {
        const unsigned short* xf = xfrag + (size_t)t * 4 * 8 * 512;
        gemm_wmma<<<dim3(1, 32), wblk, 0, stream>>>(xf, 8, hfrag, 16, encWf, b_e, gates, G4H);
        lstm_cell_enc<<<dim3(128), dim3(256), 0, stream>>>(gates, h, c, src_len, t,
                                                           enc_out, hfrag, encfrag);
    }

    // Hoisted projection: enc_proj (S*B x H) = enc_out @ W1^T ; grid (256, 8)
    gemm_wmma<<<dim3(SN * 4 / 4, 8), wblk, 0, stream>>>(encfrag, 16, nullptr, 0, W1f,
                                                        nullptr, enc_proj, HN);

    // Decoder loop
    for (int i = 0; i < SN; ++i) {
        gemm_wmma<<<dim3(1, 32), wblk, 0, stream>>>(xinfrag, 16, hfrag, 16, decWf, b_d, gates, G4H);
        lstm_cell_dec<<<dim3(128), dim3(256), 0, stream>>>(gates, h, c, hfrag);
        gemm_wmma<<<dim3(1, 8), wblk, 0, stream>>>(hfrag, 16, nullptr, 0, W2f, nullptr, hw2, HN);
        attn_score<<<dim3(2048), dim3(256), 0, stream>>>(enc_proj, hw2, vt, score);
        softmax_pack<<<dim3(BN), dim3(SN), 0, stream>>>(score, src_len, i, out_scores,
                                                        out_idx, enc_out, xinfrag);
    }
}
